// AttentivePooling_21397527069324
// MI455X (gfx1250) — matrix operand order repaired and checked by reading, not compiled, not run on hardware
//
#include <hip/hip_runtime.h>
#include <hip/hip_bf16.h>
#include <hip/hip_fp16.h>

typedef __attribute__((ext_vector_type(16))) _Float16 v16h;
typedef __attribute__((ext_vector_type(8)))  float    v8f;

#define HIDDEN 128
#define BDIM   4
#define NSEG   200
#define STOK   512

// ---------------------------------------------------------------------------
// Kernel 1: q = pad16(query) @ Wq^T + bq ; v = q @ Wc ; d[b] = q[b].bc
// Tiny real matmuls -> v_wmma_f32_16x16x32_f16, one 16-col tile per wave
// (8 waves cover all 128 output columns). Fragments use contiguous k order,
// k = 16*(lane/16) + i in both A and B (not the ISA table order). The padded
// query tile is staged in LDS so fragment builds are branch-free.
// ---------------------------------------------------------------------------
__global__ __launch_bounds__(256) void proj_kernel(
    const float* __restrict__ query, const float* __restrict__ Wq,
    const float* __restrict__ bq,    const float* __restrict__ Wc,
    const float* __restrict__ bc,    float* __restrict__ v_out,
    float* __restrict__ d_out)
{
    __shared__ _Float16 Ash[16 * 128];   // zero-padded query, f16
    __shared__ float    Qf[16 * 128];
    __shared__ _Float16 Qh[16 * 128];

    const int t    = threadIdx.x;
    const int wave = t >> 5;
    const int lane = t & 31;
    const int lo   = (lane < 16) ? 1 : 0;
    const int m    = lane & 15;      // matrix row index for this lane
    const int n0   = wave * 16;      // output column tile base

    // Stage padded A = [query ; zeros(12,128)] into LDS (uniform, no branches
    // inside the fragment-build loops).
    #pragma unroll
    for (int idx = t; idx < 16 * 128; idx += 256) {
        int row = idx >> 7;
        Ash[idx] = (row < 4) ? (_Float16)query[idx] : (_Float16)0.0f;
    }
    __syncthreads();

    // ---- Step 1: Q = A(16x128) x B, B[k][n] = Wq[n][k] ----
    v8f c = {};
    #pragma unroll
    for (int kb = 0; kb < 128; kb += 32) {
        v16h a, bfrag;
        #pragma unroll
        for (int i = 0; i < 16; ++i) {          // A: 16-bit 16x32 layout
            int g = i >> 1, p = i & 1;
            int k = kb + (lo ? 0 : 16) + i;
            a[i] = Ash[m * 128 + k];
        }
        // B: lane holds col n0+m, 16 consecutive K starting at kb + (lo?0:16)
        const float* wrow = Wq + (n0 + m) * 128 + kb + (lo ? 0 : 16);
        #pragma unroll
        for (int i = 0; i < 16; ++i) bfrag[i] = (_Float16)wrow[i];
        c = __builtin_amdgcn_wmma_f32_16x16x32_f16(false, a, false, bfrag,
                                                   (short)0, c, false, false);
    }
    // C/D layout: lane<16 -> M=r, else M=r+8 ; N = n0 + (lane&15)
    #pragma unroll
    for (int r = 0; r < 8; ++r) {
        int M = lo ? r : (r + 8);
        int Nc = n0 + m;
        float qv = c[r] + bq[Nc];               // bias (rows>=4 are don't-care)
        Qf[M * 128 + Nc] = qv;
        Qh[M * 128 + Nc] = (_Float16)qv;
    }
    __syncthreads();

    // ---- Step 2: V = Q(16x128) x Wc(128x128), B'[h][k] = Wc[h*128+k] ----
    v8f c2 = {};
    #pragma unroll
    for (int hb = 0; hb < 128; hb += 32) {
        v16h a2, b2;
        #pragma unroll
        for (int i = 0; i < 16; ++i) {
            int g = i >> 1, p = i & 1;
            int h = hb + (lo ? 0 : 16) + i;
            a2[i] = Qh[m * 128 + h];
        }
        int kcol   = n0 + m;
        int hstart = hb + (lo ? 0 : 16);
        #pragma unroll
        for (int i = 0; i < 16; ++i)
            b2[i] = (_Float16)Wc[(hstart + i) * 128 + kcol];
        c2 = __builtin_amdgcn_wmma_f32_16x16x32_f16(false, a2, false, b2,
                                                    (short)0, c2, false, false);
    }
    #pragma unroll
    for (int r = 0; r < 8; ++r) {
        int M = lo ? r : (r + 8);
        if (M < 4) v_out[M * 128 + (n0 + m)] = c2[r];
    }
    // d[b] = q[b] . bc  (tiny serial loop, 4 threads)
    if (t < 4) {
        float dd = 0.0f;
        for (int h = 0; h < 128; ++h) dd += Qf[t * 128 + h] * bc[h];
        d_out[t] = dd;
    }
}

// ---------------------------------------------------------------------------
// Kernel 2: per (b,n): att[s] = ctx[s,:].v[b] + d[b], *mask, softmax over S,
// store w to ws, result[b,n,:] = sum_s w[s]*ctx[s,:]. Block = 8 wave32s.
// ---------------------------------------------------------------------------
__global__ __launch_bounds__(256) void attn_kernel(
    const float* __restrict__ ctx, const float* __restrict__ mask,
    const float* __restrict__ v,   const float* __restrict__ dconst,
    float* __restrict__ w_out,     float* __restrict__ result)
{
    __shared__ float att[STOK];
    __shared__ __align__(16) float vsh[HIDDEN];
    __shared__ float red[16];
    __shared__ __align__(16) float part[8 * HIDDEN];

    const int t    = threadIdx.x;
    const int wave = t >> 5;
    const int lane = t & 31;
    const int bn   = blockIdx.x;          // b*NSEG + n
    const int b    = bn / NSEG;
    const float* cbase = ctx + (size_t)bn * (STOK * HIDDEN);

    if (t < HIDDEN) vsh[t] = v[b * HIDDEN + t];
    const float dv = dconst[b];
    __syncthreads();

    // ---- scores: one s-row per wave iteration, float4 coalesced ----
    const float4 vv = *(const float4*)(vsh + lane * 4);
    for (int s = wave; s < STOK; s += 8) {
        float4 cv = *(const float4*)(cbase + s * HIDDEN + lane * 4);
        float p = cv.x * vv.x + cv.y * vv.y + cv.z * vv.z + cv.w * vv.w;
        #pragma unroll
        for (int off = 16; off > 0; off >>= 1) p += __shfl_xor(p, off, 32);
        if (lane == 0) att[s] = (p + dv) * mask[(size_t)bn * STOK + s];
    }
    __syncthreads();

    // ---- softmax over 512 (multiplicative mask already applied) ----
    float lm = -3.402823466e+38f;
    for (int s = t; s < STOK; s += 256) lm = fmaxf(lm, att[s]);
    #pragma unroll
    for (int off = 16; off > 0; off >>= 1) lm = fmaxf(lm, __shfl_xor(lm, off, 32));
    if (lane == 0) red[wave] = lm;
    __syncthreads();
    if (t == 0) {
        float mm = red[0];
        #pragma unroll
        for (int i = 1; i < 8; ++i) mm = fmaxf(mm, red[i]);
        red[8] = mm;
    }
    __syncthreads();
    const float gm = red[8];
    float ls = 0.0f;
    for (int s = t; s < STOK; s += 256) {
        float e = __expf(att[s] - gm);
        att[s] = e;
        ls += e;
    }
    #pragma unroll
    for (int off = 16; off > 0; off >>= 1) ls += __shfl_xor(ls, off, 32);
    if (lane == 0) red[wave] = ls;
    __syncthreads();
    if (t == 0) {
        float ss = 0.0f;
        #pragma unroll
        for (int i = 0; i < 8; ++i) ss += red[i];
        red[9] = 1.0f / ss;
    }
    __syncthreads();
    const float inv = red[9];
    for (int s = t; s < STOK; s += 256) {
        float wv = att[s] * inv;
        att[s] = wv;
        w_out[(size_t)bn * STOK + s] = wv;
    }
    __syncthreads();

    // ---- result[b,n,:] : wave w owns s in [64w,64w+64), L2-resident reread ----
    float4 acc = {0.f, 0.f, 0.f, 0.f};
    const int sBeg = wave * 64;
    for (int s = sBeg; s < sBeg + 64; ++s) {
        float4 cv = *(const float4*)(cbase + s * HIDDEN + lane * 4);
        float wv = att[s];
        acc.x += wv * cv.x; acc.y += wv * cv.y;
        acc.z += wv * cv.z; acc.w += wv * cv.w;
    }
    *(float4*)(part + wave * HIDDEN + lane * 4) = acc;
    __syncthreads();
    if (t < HIDDEN) {
        float r = 0.0f;
        #pragma unroll
        for (int w = 0; w < 8; ++w) r += part[w * HIDDEN + t];
        result[(size_t)bn * HIDDEN + t] = r;
    }
}

// ---------------------------------------------------------------------------
// Kernel 3: token_result[b,s,:] = sum_n w[b,n,s]*ctx[b,n,s,:].
// Block owns (b, 4 consecutive s); loops n with one coalesced 2KB read per n.
// Thread t covers s = s0 + t/32, h = (t&31)*4..+3 (register accumulators).
// ---------------------------------------------------------------------------
__global__ __launch_bounds__(128) void token_kernel(
    const float* __restrict__ ctx, const float* __restrict__ w_buf,
    float* __restrict__ token)
{
    const int t     = threadIdx.x;
    const int b     = blockIdx.x >> 7;
    const int chunk = blockIdx.x & 127;
    const int s0    = chunk * 4;
    const int s     = s0 + (t >> 5);
    const float* cb = ctx   + (size_t)b * NSEG * (STOK * HIDDEN);
    const float* wb = w_buf + (size_t)b * NSEG * STOK;

    float4 acc = {0.f, 0.f, 0.f, 0.f};
    for (int n = 0; n < NSEG; ++n) {
        float wv  = wb[n * STOK + s];
        float4 cv = *(const float4*)(cb + (size_t)n * (STOK * HIDDEN)
                                        + s0 * HIDDEN + t * 4);
        acc.x += wv * cv.x; acc.y += wv * cv.y;
        acc.z += wv * cv.z; acc.w += wv * cv.w;
    }
    *(float4*)(token + ((size_t)b * STOK + s) * HIDDEN + (t & 31) * 4) = acc;
}

// ---------------------------------------------------------------------------
extern "C" void kernel_launch(void* const* d_in, const int* in_sizes, int n_in,
                              void* d_out, int out_size, void* d_ws, size_t ws_size,
                              hipStream_t stream)
{
    const float* query = (const float*)d_in[0];   // [4,128]
    const float* ctx   = (const float*)d_in[1];   // [4,200,512,128]
    const float* mask  = (const float*)d_in[2];   // [4,200,512]
    const float* Wq    = (const float*)d_in[3];   // [128,128]
    const float* bq    = (const float*)d_in[4];   // [128]
    const float* Wc    = (const float*)d_in[5];   // [128,128]
    const float* bc    = (const float*)d_in[6];   // [128]

    float* result = (float*)d_out;                          // [4,200,128]
    float* token  = (float*)d_out + BDIM * NSEG * HIDDEN;   // [4,512,128]

    float* wsf   = (float*)d_ws;
    float* v_ws  = wsf;            // 512 floats : v[b,k] = (q Wc)
    float* d_ws_ = wsf + 512;      // 4 floats   : d[b]   = q.bc
    float* w_buf = wsf + 1024;     // 409600 floats : softmax weights [4,200,512]

    proj_kernel<<<1, 256, 0, stream>>>(query, Wq, bq, Wc, bc, v_ws, d_ws_);
    attn_kernel<<<BDIM * NSEG, 256, 0, stream>>>(ctx, mask, v_ws, d_ws_,
                                                 w_buf, result);
    token_kernel<<<BDIM * 128, 128, 0, stream>>>(ctx, w_buf, token);
}
